// Point_Discriminator_11192684774175
// MI455X (gfx1250) — matrix operand order repaired and checked by reading, not compiled, not run on hardware
//
#include <hip/hip_runtime.h>

typedef __attribute__((ext_vector_type(16))) _Float16 v16h;
typedef __attribute__((ext_vector_type(8)))  float    v8f;
typedef _Float16 half_t;

#if defined(__AMDGCN__) && __has_builtin(__builtin_amdgcn_global_load_async_to_lds_b128)
#define ASYNC_LDS 1
#else
#define ASYNC_LDS 0
#endif

typedef int v4i_ __attribute__((vector_size(4 * sizeof(int))));
typedef __attribute__((address_space(1))) v4i_ as1_v4i;
typedef __attribute__((address_space(3))) v4i_ as3_v4i;

union Frag { v16h h; float4 f[2]; };

__device__ __forceinline__ float leakyf(float v, float s) { return v > 0.f ? v : s * v; }

__device__ __forceinline__ v8f vzero8() {
    v8f v;
#pragma unroll
    for (int i = 0; i < 8; ++i) v[i] = 0.f;
    return v;
}

__device__ __forceinline__ void async_wait_all() {
#if ASYNC_LDS
#if __has_builtin(__builtin_amdgcn_s_wait_asynccnt)
    __builtin_amdgcn_s_wait_asynccnt(0);
#else
    asm volatile("s_wait_asynccnt 0x0" ::: "memory");
#endif
#endif
}

// A-matrix (16x32 f16) fragment per CDNA5 layout.
// rowptr = base + row*stride + k0, where row already includes (lane&15).
__device__ __forceinline__ v16h frag_a(const half_t* rowptr) {
    int kb = ((threadIdx.x >> 4) & 1) * 8;
    Frag u;
    u.f[0] = *(const float4*)(rowptr + kb);
    u.f[1] = *(const float4*)(rowptr + 16 + kb);
    return u.h;
}

// B-matrix (32x16 f16) fragment per CDNA5 layout, from K-contiguous column storage.
// colptr = base + col*stride + k0, where col already includes (lane&15).
__device__ __forceinline__ v16h frag_b(const half_t* colptr) {
    int kb = ((threadIdx.x >> 4) & 1) * 8;
    Frag u;
    u.f[0] = *(const float4*)(colptr + kb);
    u.f[1] = *(const float4*)(colptr + 16 + kb);
    return u.h;
}

// ---------------- weight convert + pad to f16 ----------------
__global__ void k_convert(const float* __restrict__ W, half_t* __restrict__ Wo,
                          int O, int C, int Cpad) {
    int i = blockIdx.x * 256 + threadIdx.x;
    if (i >= O * Cpad) return;
    int o = i / Cpad, c = i - o * Cpad;
    float v = (c < C) ? W[(size_t)o * C + c] : 0.f;
    Wo[i] = (half_t)v;
}

// ---------------- KNN: top-17 nearest (self first) ----------------
__global__ __launch_bounds__(256) void k_knn(const float* __restrict__ points,
                                             int* __restrict__ idx, int N) {
    int b = blockIdx.y;
    int i = blockIdx.x * 256 + threadIdx.x;
    const float* px = points + (size_t)b * 3 * N;
    const float* py = px + N;
    const float* pz = py + N;
    __shared__ float sx[256], sy[256], sz[256];
    float xi = px[i], yi = py[i], zi = pz[i];
    float bd[17];
    int   bi[17];
#pragma unroll
    for (int k = 0; k < 17; ++k) { bd[k] = 3.4e38f; bi[k] = 0; }
    for (int j0 = 0; j0 < N; j0 += 256) {
        __syncthreads();
        sx[threadIdx.x] = px[j0 + threadIdx.x];
        sy[threadIdx.x] = py[j0 + threadIdx.x];
        sz[threadIdx.x] = pz[j0 + threadIdx.x];
        __syncthreads();
        for (int jj = 0; jj < 256; ++jj) {
            float dx = xi - sx[jj], dy = yi - sy[jj], dz = zi - sz[jj];
            float d = dx * dx + dy * dy + dz * dz;
            if (d < bd[16]) {
                int pos = 16;
                while (pos > 0 && bd[pos - 1] > d) {
                    bd[pos] = bd[pos - 1]; bi[pos] = bi[pos - 1]; --pos;
                }
                bd[pos] = d; bi[pos] = j0 + jj;
            }
        }
    }
    int* op = idx + ((size_t)b * N + i) * 17;
#pragma unroll
    for (int k = 0; k < 17; ++k) op[k] = bi[k];
}

// ------------- fused edge pipeline: gather -> mlp1 -> relu -> mlp2 -> kmax -------------
// 2 points per block => 34 real columns, padded to 48 (3 WMMA N-tiles).
__global__ __launch_bounds__(256) void k_edge(
    const float* __restrict__ points, const float* __restrict__ local,
    const int* __restrict__ idx,
    const half_t* __restrict__ Wef1, const float* __restrict__ bef1,
    const half_t* __restrict__ Wef2, const float* __restrict__ bef2,
    half_t* __restrict__ Fmax, int N)
{
    int b  = blockIdx.y;
    int n0 = blockIdx.x * 2;
    __shared__ __align__(16) unsigned char reg1[48 * 128 * 4]; // Fe_t (48x96 f16) then h2 (48x128 f32)
    __shared__ __align__(16) half_t h_t[48][256];
    __shared__ int   idx_s[34];
    __shared__ float selfp[3][2];
    half_t (*Fe)[96]  = (half_t(*)[96])reg1;
    float  (*h2)[128] = (float (*)[128])reg1;

    int tid = threadIdx.x;
    if (tid < 34) {
        int p = tid / 17, k = tid - p * 17;
        idx_s[tid] = idx[((size_t)b * N + (n0 + p)) * 17 + k];
    }
    if (tid < 6) {
        int d = tid >> 1, p = tid & 1;
        selfp[d][p] = points[((size_t)b * 3 + d) * N + n0 + p];
    }
    __syncthreads();

    // gather edge features transposed: Fe[col][c], c<64 local, 64..66 coords (center-rel for k>0)
    for (int e = tid; e < 48 * 96; e += 256) {
        int c = e / 48, col = e - c * 48;
        float v = 0.f;
        if (col < 34 && c < 67) {
            int p = col / 17, k = col - p * 17;
            int j = idx_s[p * 17 + k];
            if (c < 64) {
                v = local[((size_t)b * 64 + c) * N + j];
            } else {
                int d = c - 64;
                float nj = points[((size_t)b * 3 + d) * N + j];
                v = (k == 0) ? nj : (selfp[d][p] - nj);
            }
        }
        Fe[col][c] = (half_t)v;
    }
    __syncthreads();

    int lane = tid & 31, wid = tid >> 5;
    int rlane = lane & 15;
    int hi = ((lane >> 4) & 1) * 8;

    // GEMM1: (256x96) x (96x48), bias + leaky -> h_t
    for (int job = wid; job < 48; job += 8) {
        int mt = job / 3, nt = job - mt * 3;
        v8f acc = vzero8();
        const half_t* arow = Wef1 + (size_t)(mt * 16 + rlane) * 96;
        const half_t* bcol = &Fe[nt * 16 + rlane][0];
#pragma unroll
        for (int k0 = 0; k0 < 96; k0 += 32) {
            v16h a  = frag_a(arow + k0);
            v16h bb = frag_b(bcol + k0);
            acc = __builtin_amdgcn_wmma_f32_16x16x32_f16(false, a, false, bb,
                                                         (short)0, acc, false, false);
        }
        int col = nt * 16 + rlane;
#pragma unroll
        for (int r = 0; r < 8; ++r) {
            int ch = mt * 16 + r + hi;
            h_t[col][ch] = (half_t)leakyf(acc[r] + bef1[ch], 0.01f);
        }
    }
    __syncthreads();

    // GEMM2: (128x256) x (256x48), + bias -> h2 (overlays Fe region)
    for (int job = wid; job < 24; job += 8) {
        int mt = job / 3, nt = job - mt * 3;
        v8f acc = vzero8();
        const half_t* arow = Wef2 + (size_t)(mt * 16 + rlane) * 256;
        const half_t* bcol = &h_t[nt * 16 + rlane][0];
#pragma unroll
        for (int k0 = 0; k0 < 256; k0 += 32) {
            v16h a  = frag_a(arow + k0);
            v16h bb = frag_b(bcol + k0);
            acc = __builtin_amdgcn_wmma_f32_16x16x32_f16(false, a, false, bb,
                                                         (short)0, acc, false, false);
        }
        int col = nt * 16 + rlane;
#pragma unroll
        for (int r = 0; r < 8; ++r) {
            int ch = mt * 16 + r + hi;
            h2[col][ch] = acc[r] + bef2[ch];
        }
    }
    __syncthreads();

    // max over the 17 neighbor slots per point
    {
        int ch = tid & 127, p = tid >> 7;
        float m = -3.4e38f;
#pragma unroll
        for (int k = 0; k < 17; ++k) {
            float v = h2[p * 17 + k][ch];
            m = v > m ? v : m;
        }
        Fmax[((size_t)b * 128 + ch) * N + n0 + p] = (half_t)m;
    }
}

// ---------------- concat [g | local | fmax | xyz | pad] -> f16 ----------------
__global__ void k_concat(const float* __restrict__ g, const float* __restrict__ loc,
                         const half_t* __restrict__ Fmax, const float* __restrict__ pts,
                         half_t* __restrict__ X, int N, int B) {
    size_t i = (size_t)blockIdx.x * 256 + threadIdx.x;
    size_t per = (size_t)1248 * N;
    if (i >= (size_t)B * per) return;
    int b = (int)(i / per);
    size_t rem = i - (size_t)b * per;
    int c = (int)(rem / N);
    int n = (int)(rem - (size_t)c * N);
    half_t v;
    if (c < 1024)      v = (half_t)g[(size_t)b * 1024 + c];
    else if (c < 1088) v = (half_t)loc[((size_t)b * 64 + (c - 1024)) * N + n];
    else if (c < 1216) v = Fmax[((size_t)b * 128 + (c - 1088)) * N + n];
    else if (c < 1219) v = (half_t)pts[((size_t)b * 3 + (c - 1216)) * N + n];
    else               v = (half_t)0.f;
    X[i] = v;
}

// -------- stage one K-step tile pair into LDS (W layout-preserving, X transposed) --------
__device__ __forceinline__ void stage_tiles(
    const half_t* __restrict__ W, const half_t* __restrict__ Xb,
    half_t (*Ws)[32], half_t (*Xt)[32],
    int o0, int m0, int k0, int Cpad, int M, int tid)
{
    // W tile: 128 rows x 32 halves, 16B chunks. Async DMA to LDS when available.
#if ASYNC_LDS
    for (int e = tid; e < 512; e += 256) {
        int row = e >> 2, koff = (e & 3) * 8;
        const half_t* gp = W + (size_t)(o0 + row) * Cpad + k0 + koff;
        half_t* lp = &Ws[row][koff];
        __builtin_amdgcn_global_load_async_to_lds_b128((as1_v4i*)gp, (as3_v4i*)lp, 0, 0);
    }
#else
    for (int e = tid; e < 512; e += 256) {
        int row = e >> 2, koff = (e & 3) * 8;
        *(float4*)&Ws[row][koff] =
            *(const float4*)(W + (size_t)(o0 + row) * Cpad + k0 + koff);
    }
#endif
    // X tile: read 8 columns per b128 load, scatter-transpose into Xt[col][kk]
    for (int e = tid; e < 512; e += 256) {
        int kk = e >> 4, colg = (e & 15) * 8;
        float4 d = *(const float4*)(Xb + (size_t)(k0 + kk) * M + m0 + colg);
        const half_t* hv = (const half_t*)&d;
#pragma unroll
        for (int j = 0; j < 8; ++j) Xt[colg + j][kk] = hv[j];
    }
}

// ---------------- double-buffered WMMA GEMM: Y = leaky(W*X + bias) ----------------
// W: [O][Cpad] f16, X: [B][Cpad][M] f16, Y: [B][O][M] f16. O%128==0, Cpad%32==0, M%128==0.
// Block tile 128x128, 8 waves each owning 32x64 (2x4 accumulators, 8 WMMA / K-step).
__global__ __launch_bounds__(256) void k_gemm(
    const half_t* __restrict__ W, const half_t* __restrict__ X,
    half_t* __restrict__ Y, const float* __restrict__ bias,
    int O, int Cpad, int M, float slope)
{
    int bz = blockIdx.z;
    const half_t* Xb = X + (size_t)bz * Cpad * M;
    half_t*       Yb = Y + (size_t)bz * O * M;
    int o0 = blockIdx.x * 128;
    int m0 = blockIdx.y * 128;
    __shared__ __align__(16) half_t Ws[2][128][32];
    __shared__ __align__(16) half_t Xt[2][128][32];

    int tid = threadIdx.x, lane = tid & 31, wid = tid >> 5;
    int rlane = lane & 15;
    int hi = ((lane >> 4) & 1) * 8;
    int wR = wid >> 1, wC = wid & 1;

    v8f acc[2][4];
#pragma unroll
    for (int a = 0; a < 2; ++a)
#pragma unroll
        for (int c = 0; c < 4; ++c) acc[a][c] = vzero8();

    int T = Cpad / 32;
    stage_tiles(W, Xb, Ws[0], Xt[0], o0, m0, 0, Cpad, M, tid);
    async_wait_all();
    __syncthreads();

    for (int t = 0; t < T; ++t) {
        int cur = t & 1;
        if (t + 1 < T)
            stage_tiles(W, Xb, Ws[cur ^ 1], Xt[cur ^ 1], o0, m0, (t + 1) * 32, Cpad, M, tid);

        v16h a0 = frag_a(&Ws[cur][wR * 32 + rlane][0]);
        v16h a1 = frag_a(&Ws[cur][wR * 32 + 16 + rlane][0]);
#pragma unroll
        for (int ct = 0; ct < 4; ++ct) {
            v16h bf = frag_b(&Xt[cur][wC * 64 + ct * 16 + rlane][0]);
            acc[0][ct] = __builtin_amdgcn_wmma_f32_16x16x32_f16(false, a0, false, bf,
                                                                (short)0, acc[0][ct], false, false);
            acc[1][ct] = __builtin_amdgcn_wmma_f32_16x16x32_f16(false, a1, false, bf,
                                                                (short)0, acc[1][ct], false, false);
        }
        if (t + 1 < T) {
            async_wait_all();
            __syncthreads();
        }
    }

#pragma unroll
    for (int rt = 0; rt < 2; ++rt)
#pragma unroll
        for (int ct = 0; ct < 4; ++ct) {
            int col = m0 + wC * 64 + ct * 16 + rlane;
#pragma unroll
            for (int r = 0; r < 8; ++r) {
                int row = o0 + wR * 32 + rt * 16 + r + hi;
                float v = acc[rt][ct][r] + bias[row];
                v = v > 0.f ? v : slope * v;
                Yb[(size_t)row * M + col] = (half_t)v;
            }
        }
}

// ---------------- final 1x256 projection (fp32 out) ----------------
__global__ void k_final(const half_t* __restrict__ Y2, const float* __restrict__ W3,
                        const float* __restrict__ b3, float* __restrict__ out,
                        int N, int B) {
    int i = blockIdx.x * 256 + threadIdx.x;
    if (i >= B * N) return;
    int b = i / N, n = i - b * N;
    const half_t* y = Y2 + (size_t)b * 256 * N + n;
    float acc = b3[0];
    for (int c = 0; c < 256; ++c)
        acc += W3[c] * (float)y[(size_t)c * N];
    out[i] = acc;
}

extern "C" void kernel_launch(void* const* d_in, const int* in_sizes, int n_in,
                              void* d_out, int out_size, void* d_ws, size_t ws_size,
                              hipStream_t stream) {
    const float* g     = (const float*)d_in[0];
    const float* pts   = (const float*)d_in[1];
    const float* loc   = (const float*)d_in[2];
    const float* Wef1f = (const float*)d_in[3];
    const float* bef1  = (const float*)d_in[4];
    const float* Wef2f = (const float*)d_in[5];
    const float* bef2  = (const float*)d_in[6];
    const float* W1f   = (const float*)d_in[7];
    const float* b1    = (const float*)d_in[8];
    const float* W2f   = (const float*)d_in[9];
    const float* b2    = (const float*)d_in[10];
    const float* W3    = (const float*)d_in[11];
    const float* b3    = (const float*)d_in[12];
    float* out = (float*)d_out;

    int B = in_sizes[0] / 1024;
    int N = in_sizes[1] / (3 * B);
    (void)n_in; (void)out_size; (void)ws_size;

    unsigned char* w = (unsigned char*)d_ws;
    size_t off = 0;
    auto alloc = [&](size_t bytes) -> void* {
        void* p = (void*)(w + off);
        off = (off + bytes + 255) & ~(size_t)255;
        return p;
    };
    int*    idx   = (int*)   alloc((size_t)B * N * 17 * 4);
    half_t* Wef1h = (half_t*)alloc((size_t)256 * 96 * 2);
    half_t* Wef2h = (half_t*)alloc((size_t)128 * 256 * 2);
    half_t* W1h   = (half_t*)alloc((size_t)1024 * 1248 * 2);
    half_t* W2h   = (half_t*)alloc((size_t)256 * 1024 * 2);
    half_t* Fmax  = (half_t*)alloc((size_t)B * 128 * N * 2);
    half_t* Xf    = (half_t*)alloc((size_t)B * 1248 * N * 2);
    half_t* Y1    = (half_t*)alloc((size_t)B * 1024 * N * 2);
    half_t* Y2    = (half_t*)alloc((size_t)B * 256 * N * 2);

    k_convert<<<(256 * 96 + 255) / 256, 256, 0, stream>>>(Wef1f, Wef1h, 256, 67, 96);
    k_convert<<<(128 * 256 + 255) / 256, 256, 0, stream>>>(Wef2f, Wef2h, 128, 256, 256);
    k_convert<<<(1024 * 1248 + 255) / 256, 256, 0, stream>>>(W1f, W1h, 1024, 1219, 1248);
    k_convert<<<(256 * 1024 + 255) / 256, 256, 0, stream>>>(W2f, W2h, 256, 1024, 1024);

    k_knn<<<dim3(N / 256, B), 256, 0, stream>>>(pts, idx, N);

    k_edge<<<dim3(N / 2, B), 256, 0, stream>>>(pts, loc, idx, Wef1h, bef1, Wef2h, bef2, Fmax, N);

    {
        size_t tot = (size_t)B * 1248 * N;
        k_concat<<<(unsigned)((tot + 255) / 256), 256, 0, stream>>>(g, loc, Fmax, pts, Xf, N, B);
    }

    k_gemm<<<dim3(1024 / 128, N / 128, B), 256, 0, stream>>>(W1h, Xf, Y1, b1, 1024, 1248, N, 0.01f);
    k_gemm<<<dim3(256 / 128, N / 128, B), 256, 0, stream>>>(W2h, Y1, Y2, b2, 256, 1024, N, 0.01f);

    k_final<<<(B * N + 255) / 256, 256, 0, stream>>>(Y2, W3, b3, out, N, B);
}
